// MixtureOfExperts_78159814852666
// MI455X (gfx1250) — matrix operand order repaired and checked by reading, not compiled, not run on hardware
//
#include <hip/hip_runtime.h>
#include <hip/hip_bf16.h>

// ---------------- problem constants (from reference) ----------------
#define MOE_B 4
#define MOE_S 2048
#define MOE_D 1024
#define MOE_H 4096
#define MOE_E 8
#define MOE_K 2
#define MOE_N (MOE_B * MOE_S)   // 8192 tokens
#define MOE_NK (MOE_N * MOE_K)  // 16384 (token, slot) pairs
#define MOE_C 3072              // capacity per expert

// ---------------- CDNA5 async global->LDS copy (guarded) ----------------
#if defined(__has_builtin)
#if __has_builtin(__builtin_amdgcn_global_load_async_to_lds_b128) && \
    __has_builtin(__builtin_amdgcn_s_wait_asynccnt)
#define MOE_ASYNC_COPY 1
#endif
#endif
#ifndef MOE_ASYNC_COPY
#define MOE_ASYNC_COPY 0
#endif

// ---------------- WMMA types (gfx1250, wave32) ----------------
typedef __bf16 bf16_t;
typedef bf16_t bf16x16 __attribute__((ext_vector_type(16)));
typedef float floatx8 __attribute__((ext_vector_type(8)));
typedef unsigned int uint4v __attribute__((ext_vector_type(4)));

union Frag {
  bf16x16 v;
  unsigned u[8];
};

__device__ __forceinline__ unsigned short f2bf(float f) {
  // round-to-nearest-even float32 -> bf16
  unsigned u = __float_as_uint(f);
  u += 0x7FFFu + ((u >> 16) & 1u);
  return (unsigned short)(u >> 16);
}

#if MOE_ASYNC_COPY
// Builtin parameter pointee type (from clang diagnostic): GCC-vector v4i.
typedef int v4i_gcc __attribute__((vector_size(4 * sizeof(int))));
typedef __attribute__((address_space(1))) v4i_gcc* moe_gv4p;  // global
typedef __attribute__((address_space(3))) v4i_gcc* moe_lv4p;  // LDS

__device__ __forceinline__ void moe_async_b128(const unsigned short* g,
                                               unsigned short* l) {
  // AMDGPU: flat LDS addresses carry the LDS offset in the low 32 bits, and
  // flat global addresses equal AS1 addresses, so integer casts are exact.
  // AS-qualified pointers also implicitly convert to generic if the builtin
  // is declared with generic params.
  __builtin_amdgcn_global_load_async_to_lds_b128(
      (moe_gv4p)(unsigned long long)g, (moe_lv4p)(unsigned long long)l,
      /*offset=*/0, /*cpol=*/0);
}
#endif

// ---------------- 1) weight transpose + fp32->bf16 convert ----------------
// In:  per-expert [R][Cc] fp32 (row-major).  Out: per-expert [Cc][R] bf16.
__global__ __launch_bounds__(256) void moe_transpose_convert(
    const float* __restrict__ In, unsigned short* __restrict__ Out,
    int R, int Cc) {
  __shared__ float tile[32][33];
  const int e = blockIdx.z;
  const float* in = In + (size_t)e * R * Cc;
  unsigned short* out = Out + (size_t)e * R * Cc;
  const int c0 = blockIdx.x * 32;
  const int r0 = blockIdx.y * 32;
  const int tx = threadIdx.x & 31;
  const int ty = threadIdx.x >> 5;  // 0..7
#pragma unroll
  for (int i = 0; i < 32; i += 8)
    tile[ty + i][tx] = in[(size_t)(r0 + ty + i) * Cc + (c0 + tx)];
  __syncthreads();
#pragma unroll
  for (int i = 0; i < 32; i += 8)
    out[(size_t)(c0 + ty + i) * R + (r0 + tx)] = f2bf(tile[tx][ty + i]);
}

// ---------------- 2) router: logits -> softmax -> top-2 -> renorm ----------------
__global__ __launch_bounds__(256) void moe_router(
    const float* __restrict__ x, const float* __restrict__ wr,
    int* __restrict__ eidx, float* __restrict__ gate) {
  __shared__ float wsh[MOE_E * MOE_D];  // 32 KB, stored [e][d] (conflict-free)
  for (int i = threadIdx.x; i < MOE_E * MOE_D; i += 256)
    wsh[(i & 7) * MOE_D + (i >> 3)] = wr[i];
  __syncthreads();

  const int lane = threadIdx.x & 31;
  const int wave = threadIdx.x >> 5;
  const int n = blockIdx.x * 8 + wave;  // one wave per token
  const float* xr = x + (size_t)n * MOE_D;

  float acc[MOE_E];
#pragma unroll
  for (int e = 0; e < MOE_E; ++e) acc[e] = 0.f;
  for (int d = lane; d < MOE_D; d += 32) {
    const float xv = xr[d];
#pragma unroll
    for (int e = 0; e < MOE_E; ++e) acc[e] += xv * wsh[e * MOE_D + d];
  }
#pragma unroll
  for (int e = 0; e < MOE_E; ++e) {
#pragma unroll
    for (int off = 16; off > 0; off >>= 1) acc[e] += __shfl_xor(acc[e], off, 32);
  }
  if (lane == 0) {
    float mx = acc[0];
#pragma unroll
    for (int e = 1; e < MOE_E; ++e) mx = fmaxf(mx, acc[e]);
    float p[MOE_E];
    float s = 0.f;
#pragma unroll
    for (int e = 0; e < MOE_E; ++e) {
      p[e] = __expf(acc[e] - mx);
      s += p[e];
    }
    const float inv = 1.f / s;
#pragma unroll
    for (int e = 0; e < MOE_E; ++e) p[e] *= inv;
    int i0 = 0;
#pragma unroll
    for (int e = 1; e < MOE_E; ++e)
      if (p[e] > p[i0]) i0 = e;  // ties -> lowest index, matches top_k
    int i1 = (i0 == 0) ? 1 : 0;
#pragma unroll
    for (int e = 0; e < MOE_E; ++e)
      if (e != i0 && e != i1 && p[e] > p[i1]) i1 = e;
    const float g0 = p[i0], g1 = p[i1];
    const float r = 1.f / (g0 + g1);
    eidx[2 * n + 0] = i0;
    eidx[2 * n + 1] = i1;
    gate[2 * n + 0] = g0 * r;
    gate[2 * n + 1] = g1 * r;
  }
}

// ---------------- 3) capacity: ordered per-expert running count ----------------
// Single workgroup; reproduces reference's token-major (slot-fastest) cumsum exactly.
__global__ __launch_bounds__(256) void moe_capacity(
    const int* __restrict__ eidx, float* __restrict__ gate,
    int* __restrict__ pos, int* __restrict__ counts) {
  __shared__ int wcnt[8][MOE_E];
  __shared__ int woff[8][MOE_E];
  __shared__ int base[MOE_E];
  const int lane = threadIdx.x & 31;
  const int wave = threadIdx.x >> 5;
  if (threadIdx.x < MOE_E) base[threadIdx.x] = 0;
  __syncthreads();

  for (int i0 = 0; i0 < MOE_NK; i0 += 256) {
    const int i = i0 + threadIdx.x;
    const int my_e = eidx[i];
    const unsigned lt = (1u << lane) - 1u;
    int my_pref = 0;
#pragma unroll
    for (int e = 0; e < MOE_E; ++e) {
      const unsigned m = (unsigned)__ballot(my_e == e);
      if (lane == 0) wcnt[wave][e] = __popc(m);
      if (my_e == e) my_pref = __popc(m & lt);
    }
    __syncthreads();
    if (threadIdx.x < MOE_E) {  // thread t scans waves for expert t
      int off = base[threadIdx.x];
#pragma unroll
      for (int w = 0; w < 8; ++w) {
        woff[w][threadIdx.x] = off;
        off += wcnt[w][threadIdx.x];
      }
      base[threadIdx.x] = off;
    }
    __syncthreads();
    const int p = woff[wave][my_e] + my_pref;
    const bool keep = p < MOE_C;
    pos[i] = keep ? p : 0;
    if (!keep) gate[i] = 0.f;  // mask dropped tokens
    __syncthreads();
  }
  if (threadIdx.x < MOE_E) {
    const int c = base[threadIdx.x];
    counts[threadIdx.x] = c < MOE_C ? c : MOE_C;
  }
}

// ---------------- 4) dispatch: scatter token rows -> bf16 expert buffers ----------------
__global__ __launch_bounds__(128) void moe_dispatch(
    const float* __restrict__ x, const int* __restrict__ eidx,
    const int* __restrict__ pos, const float* __restrict__ gate,
    unsigned short* __restrict__ Abf) {
  const int i = blockIdx.x;      // (token, slot) pair
  if (!(gate[i] > 0.f)) return;  // dropped -> contributes nothing
  const int n = i >> 1;
  const int e = eidx[i];
  const int p = pos[i];
  const float* src = x + (size_t)n * MOE_D;
  unsigned short* dst = Abf + ((size_t)e * MOE_C + p) * MOE_D;
  for (int d = threadIdx.x; d < MOE_D; d += 128) dst[d] = f2bf(src[d]);
}

// ---------------- 5/6) grouped GEMM via v_wmma_f32_16x16x32_bf16 ----------------
// A:  [E][M][Kd]  bf16 row-major (K contiguous)
// Bt: [E][Nd][Kd] bf16 N-major   (K contiguous)  => Out = A * Bt^T
//
// Block tile 128(M) x 128(N), double-buffered LDS staging, K-step 32.
// 8 waves in a 4(M) x 2(N) grid; each wave owns a 32x64 subtile = 8 WMMAs/step.
// Staging uses CDNA5 async global->LDS copies (ASYNCcnt) when available,
// else global->VGPR->ds_store. LDS rows padded to 40 halves (80 B).
#define LDS_PITCH 40

template <bool RELU_BF16OUT>
__global__ __launch_bounds__(256) void moe_grouped_gemm(
    const unsigned short* __restrict__ A, const unsigned short* __restrict__ Bt,
    const float* __restrict__ bias, void* __restrict__ OutP,
    const int* __restrict__ counts, int M, int Kd, int Nd) {
  __shared__ __align__(16) unsigned short Atile[2][128 * LDS_PITCH];
  __shared__ __align__(16) unsigned short Btile[2][128 * LDS_PITCH];

  const int e = blockIdx.z;
  const int m0blk = blockIdx.y * 128;
  if (m0blk >= counts[e]) return;  // uniform: whole block skips empty slots
  const int n0blk = blockIdx.x * 128;

  const int tid = threadIdx.x;
  const int wave = tid >> 5;
  const int lane = tid & 31;
  const int wm = wave & 3;   // M subtile (32 rows)
  const int wn = wave >> 2;  // N subtile (64 cols)
  const int half = lane >> 4;
  const int l15 = lane & 15;

  // ---- staging: thread t loads row (t>>1), 16-half chunk (t&1) of each tile
  const int sr = tid >> 1;
  const int scH = (tid & 1) << 4;  // 0 or 16 halves
  const unsigned short* ag = A + ((size_t)e * M + (m0blk + sr)) * (size_t)Kd + scH;
  const unsigned short* bg = Bt + ((size_t)e * Nd + (n0blk + sr)) * (size_t)Kd + scH;
  const int ldsti = sr * LDS_PITCH + scH;

  // ---- accumulators: 2 (M sub) x 4 (N sub) WMMA tiles
  floatx8 acc0[4], acc1[4];
#pragma unroll
  for (int j = 0; j < 4; ++j)
#pragma unroll
    for (int v = 0; v < 8; ++v) {
      acc0[j][v] = 0.f;
      acc1[j][v] = 0.f;
    }

  // ---- prologue: stage K-step 0 into buffer 0
#if MOE_ASYNC_COPY
  moe_async_b128(ag, &Atile[0][ldsti]);
  moe_async_b128(ag + 8, &Atile[0][ldsti + 8]);
  moe_async_b128(bg, &Btile[0][ldsti]);
  moe_async_b128(bg + 8, &Btile[0][ldsti + 8]);
  __builtin_amdgcn_s_wait_asynccnt(0);
#else
  {
    uint4v a0 = *(const uint4v*)(ag);
    uint4v a1 = *(const uint4v*)(ag + 8);
    uint4v b0 = *(const uint4v*)(bg);
    uint4v b1 = *(const uint4v*)(bg + 8);
    *(uint4v*)&Atile[0][ldsti] = a0;
    *(uint4v*)&Atile[0][ldsti + 8] = a1;
    *(uint4v*)&Btile[0][ldsti] = b0;
    *(uint4v*)&Btile[0][ldsti + 8] = b1;
  }
#endif
  __syncthreads();

  for (int kk = 0; kk < Kd; kk += 32) {
    const int p = (kk >> 5) & 1;
    const bool hasNext = (kk + 32) < Kd;

    // ---- issue staging for the next K-step early (latency overlaps WMMAs);
    //      buffer p^1 is free: the barrier ending the previous step ensured
    //      every wave finished reading it.
#if MOE_ASYNC_COPY
    if (hasNext) {
      const int q = p ^ 1;
      moe_async_b128(ag + kk + 32, &Atile[q][ldsti]);
      moe_async_b128(ag + kk + 40, &Atile[q][ldsti + 8]);
      moe_async_b128(bg + kk + 32, &Btile[q][ldsti]);
      moe_async_b128(bg + kk + 40, &Btile[q][ldsti + 8]);
    }
#else
    uint4v na0, na1, nb0, nb1;
    if (hasNext) {
      na0 = *(const uint4v*)(ag + kk + 32);
      na1 = *(const uint4v*)(ag + kk + 40);
      nb0 = *(const uint4v*)(bg + kk + 32);
      nb1 = *(const uint4v*)(bg + kk + 40);
    }
#endif

    // ---- compute from LDS buffer p
    // A fragment (ISA 7.12.2, 16-bit A 16x32): lane = M row; 16B chunks at
    // half*8 and half*8+16 halves.
    Frag a0, a1;
    {
      const unsigned short* ab0 =
          &Atile[p][(wm * 32 + l15) * LDS_PITCH + half * 8];
      const unsigned short* ab1 = ab0 + 16 * LDS_PITCH;
      *(uint4v*)&a0.u[0] = *(const uint4v*)(ab0);
      *(uint4v*)&a0.u[4] = *(const uint4v*)(ab0 + 16);
      *(uint4v*)&a1.u[0] = *(const uint4v*)(ab1);
      *(uint4v*)&a1.u[4] = *(const uint4v*)(ab1 + 16);
    }
#pragma unroll
    for (int j = 0; j < 4; ++j) {
      // B fragment (16-bit B 32x16): lane = N col; 16B chunks at
      // half*8 and half*8+16 halves.
      Frag b;
      const unsigned short* bb =
          &Btile[p][(wn * 64 + 16 * j + l15) * LDS_PITCH + half * 8];
      *(uint4v*)&b.u[0] = *(const uint4v*)(bb);
      *(uint4v*)&b.u[4] = *(const uint4v*)(bb + 16);
      acc0[j] = __builtin_amdgcn_wmma_f32_16x16x32_bf16(
          false, a0.v, false, b.v, (short)0, acc0[j], false, false);
      acc1[j] = __builtin_amdgcn_wmma_f32_16x16x32_bf16(
          false, a1.v, false, b.v, (short)0, acc1[j], false, false);
    }

    // ---- make next buffer visible before the barrier
#if MOE_ASYNC_COPY
    if (hasNext) __builtin_amdgcn_s_wait_asynccnt(0);
#else
    if (hasNext) {
      const int q = p ^ 1;
      *(uint4v*)&Atile[q][ldsti] = na0;
      *(uint4v*)&Atile[q][ldsti + 8] = na1;
      *(uint4v*)&Btile[q][ldsti] = nb0;
      *(uint4v*)&Btile[q][ldsti + 8] = nb1;
    }
#endif
    __syncthreads();
  }

  // ---- epilogue: C/D layout -> M = 8*half + v (+16 for second subtile),
  //      N = 16*j + l15; fused bias (+ ReLU + bf16 pack for GEMM1)
  const int mb0 = m0blk + wm * 32 + half * 8;
  const int mb1 = mb0 + 16;
#pragma unroll
  for (int j = 0; j < 4; ++j) {
    const int n = n0blk + wn * 64 + 16 * j + l15;
    const float bv = bias[(size_t)e * Nd + n];
#pragma unroll
    for (int v = 0; v < 8; ++v) {
      float r0 = acc0[j][v] + bv;
      float r1 = acc1[j][v] + bv;
      const size_t i0 = ((size_t)e * M + (mb0 + v)) * (size_t)Nd + n;
      const size_t i1 = ((size_t)e * M + (mb1 + v)) * (size_t)Nd + n;
      if (RELU_BF16OUT) {
        r0 = r0 > 0.f ? r0 : 0.f;
        r1 = r1 > 0.f ? r1 : 0.f;
        ((unsigned short*)OutP)[i0] = f2bf(r0);
        ((unsigned short*)OutP)[i1] = f2bf(r1);
      } else {
        ((float*)OutP)[i0] = r0;
        ((float*)OutP)[i1] = r1;
      }
    }
  }
}

// ---------------- 7) combine: gather + gate-weighted sum ----------------
__global__ __launch_bounds__(256) void moe_combine(
    const float* __restrict__ Obuf, const int* __restrict__ eidx,
    const int* __restrict__ pos, const float* __restrict__ gate,
    float* __restrict__ y) {
  const int n = blockIdx.x;
  const int e0 = eidx[2 * n + 0], e1 = eidx[2 * n + 1];
  const int p0 = pos[2 * n + 0], p1 = pos[2 * n + 1];
  const float g0 = gate[2 * n + 0], g1 = gate[2 * n + 1];
  const float* r0 = Obuf + ((size_t)e0 * MOE_C + p0) * MOE_D;
  const float* r1 = Obuf + ((size_t)e1 * MOE_C + p1) * MOE_D;
  float* yr = y + (size_t)n * MOE_D;
  for (int d = threadIdx.x; d < MOE_D; d += 256)
    yr[d] = g0 * r0[d] + g1 * r1[d];
}

// ---------------- launch ----------------
extern "C" void kernel_launch(void* const* d_in, const int* in_sizes, int n_in,
                              void* d_out, int out_size, void* d_ws,
                              size_t ws_size, hipStream_t stream) {
  const float* x = (const float*)d_in[0];         // [B,S,D]
  const float* w_router = (const float*)d_in[1];  // [D,E]
  const float* w1 = (const float*)d_in[2];        // [E,D,H]
  const float* b1 = (const float*)d_in[3];        // [E,H]
  const float* w2 = (const float*)d_in[4];        // [E,H,D]
  const float* b2 = (const float*)d_in[5];        // [E,D]
  float* y = (float*)d_out;                       // [B,S,D]

  char* ws = (char*)d_ws;
  size_t off = 0;
  auto take = [&](size_t bytes) -> void* {
    void* p = ws + off;
    off = (off + bytes + 255) & ~(size_t)255;
    return p;
  };
  int* eidx = (int*)take((size_t)MOE_NK * sizeof(int));
  float* gate = (float*)take((size_t)MOE_NK * sizeof(float));
  int* pos = (int*)take((size_t)MOE_NK * sizeof(int));
  int* counts = (int*)take((size_t)MOE_E * sizeof(int));
  unsigned short* w1t = (unsigned short*)take((size_t)MOE_E * MOE_H * MOE_D * 2);
  unsigned short* w2t = (unsigned short*)take((size_t)MOE_E * MOE_D * MOE_H * 2);
  unsigned short* Abf = (unsigned short*)take((size_t)MOE_E * MOE_C * MOE_D * 2);
  unsigned short* Hbuf = (unsigned short*)take((size_t)MOE_E * MOE_C * MOE_H * 2);
  float* Obuf = (float*)take((size_t)MOE_E * MOE_C * MOE_D * sizeof(float));
  (void)ws_size;
  (void)in_sizes;
  (void)n_in;
  (void)out_size;

  const dim3 blk256(256);
  // w1 [E][D][H] -> w1t [E][H][D] bf16 ; w2 [E][H][D] -> w2t [E][D][H] bf16
  moe_transpose_convert<<<dim3(MOE_H / 32, MOE_D / 32, MOE_E), blk256, 0,
                          stream>>>(w1, w1t, MOE_D, MOE_H);
  moe_transpose_convert<<<dim3(MOE_D / 32, MOE_H / 32, MOE_E), blk256, 0,
                          stream>>>(w2, w2t, MOE_H, MOE_D);
  moe_router<<<MOE_N / 8, blk256, 0, stream>>>(x, w_router, eidx, gate);
  moe_capacity<<<1, blk256, 0, stream>>>(eidx, gate, pos, counts);
  moe_dispatch<<<MOE_NK, dim3(128), 0, stream>>>(x, eidx, pos, gate, Abf);
  // GEMM1: [C,D] x [D,H] -> relu -> bf16 [C,H], per expert
  moe_grouped_gemm<true><<<dim3(MOE_H / 128, MOE_C / 128, MOE_E), blk256, 0,
                           stream>>>(Abf, w1t, b1, (void*)Hbuf, counts, MOE_C,
                                     MOE_D, MOE_H);
  // GEMM2: [C,H] x [H,D] -> f32 [C,D], per expert
  moe_grouped_gemm<false><<<dim3(MOE_D / 128, MOE_C / 128, MOE_E), blk256, 0,
                            stream>>>(Hbuf, w2t, b2, (void*)Obuf, counts, MOE_C,
                                      MOE_H, MOE_D);
  moe_combine<<<MOE_N, blk256, 0, stream>>>(Obuf, eidx, pos, gate, y);
}
